// MeanEstimationBaseline_83305185673687
// MI455X (gfx1250) — matrix operand order repaired and checked by reading, not compiled, not run on hardware
//
#include <hip/hip_runtime.h>
#include <hip/hip_bf16.h>
#include <stdint.h>

// MI455X / gfx1250, wave32 WMMA path.
// Stage 0: swizzle W once into the per-lane WMMA B-matrix layout (16 KB table).
// Stage 1: fT[s][col][b] = bf16( (X_s @ W + b)[b][col] )   (transposed, bf16, L2-resident)
// Stage 2: A[i, j*64+k] = sum_b fxT[i][b] * (fyT[j][b]*fzT[k][b])  via v_wmma_f32_16x16x32_bf16,
//          with the B-matrix formed in-register by v_pk_mul_bf16 (never materialized).
// Stage 3: reduce 8 K-split partials (stored in C-native layout), scale by 1/B.

typedef __attribute__((ext_vector_type(16))) __bf16   v16bf;
typedef __attribute__((ext_vector_type(8)))  float    v8f;
typedef __attribute__((ext_vector_type(4)))  float    f4;
typedef __attribute__((ext_vector_type(4)))  uint32_t u4;

#define BTOT  65536
#define INDIM 128
#define DDIM  64

union BV {
  v16bf    v;
  uint32_t u[8];
  __bf16   h[16];
};

static __device__ inline uint32_t pk_mul_bf16(uint32_t a, uint32_t b) {
  uint32_t d;
  asm volatile("v_pk_mul_bf16 %0, %1, %2" : "=v"(d) : "v"(a), "v"(b));
  return d;
}

static __device__ inline uint32_t pack2_bf16(float lo, float hi) {
  union { __bf16 h[2]; uint32_t u; } p;
  p.h[0] = (__bf16)lo;
  p.h[1] = (__bf16)hi;
  return p.u;
}

// ---------------- Kernel 0: swizzle W into WMMA B-operand layout ----------------
// Wsw[tile][lane][e], tile = (kb/32)*4 + ct  (16 tiles of 32x16)
// lane holds col = ct*16 + lane%16; element e <-> K = kb + 8*(lane/16) + e%8 + 16*(e/8)
__global__ __launch_bounds__(256) void wswz_kernel(
    const float* __restrict__ W, __bf16* __restrict__ Wsw) {
  const int idx  = blockIdx.x * 256 + threadIdx.x;  // 0..8191
  const int tile = idx >> 9;
  const int lane = (idx >> 4) & 31;
  const int e    = idx & 15;
  const int kb   = (tile >> 2) * 32;
  const int ct   = tile & 3;
  const int K    = kb + 8 * (lane >> 4) + (e & 7) + 16 * (e >> 3);
  const int col  = ct * 16 + (lane & 15);
  Wsw[idx] = (__bf16)W[K * DDIM + col];
}

// ---------------- Kernel 1: projection + transpose-store (bf16) ----------------
__global__ __launch_bounds__(256) void proj_kernel(
    const float* __restrict__ x, const float* __restrict__ y,
    const float* __restrict__ z, const __bf16* __restrict__ Wsw,
    const float* __restrict__ bias, __bf16* __restrict__ fT) {
  const int tid  = threadIdx.x;
  const int which = blockIdx.y;  // 0=x, 1=y, 2=z
  const float* __restrict__ src = (which == 0) ? x : (which == 1) ? y : z;
  __bf16* __restrict__ dst = fT + (size_t)which * DDIM * BTOT;

  const int wave = tid >> 5;
  const int lane = tid & 31;
  const int lr   = lane & 15;
  const int g    = lane >> 4;

  const int m0  = blockIdx.x * 128 + wave * 16;  // 16 batch rows per wave
  const int row = m0 + lr;

  v8f acc[4] = {};
  const float* __restrict__ rowp = src + (size_t)row * INDIM;
  const __bf16* __restrict__ wl = Wsw + lane * 16;  // this lane's slice of each tile

#pragma unroll
  for (int kb = 0; kb < INDIM; kb += 32) {
    // A tile 16x32: element e <-> K = (e/8)*16 + 8g + e%8  -> two runs of 8 floats
    BV a;
    f4 r0 = *reinterpret_cast<const f4*>(rowp + kb + 8 * g);
    f4 r1 = *reinterpret_cast<const f4*>(rowp + kb + 8 * g + 4);
    f4 r2 = *reinterpret_cast<const f4*>(rowp + kb + 16 + 8 * g);
    f4 r3 = *reinterpret_cast<const f4*>(rowp + kb + 16 + 8 * g + 4);
    a.u[0] = pack2_bf16(r0.x, r0.y); a.u[1] = pack2_bf16(r0.z, r0.w);
    a.u[2] = pack2_bf16(r1.x, r1.y); a.u[3] = pack2_bf16(r1.z, r1.w);
    a.u[4] = pack2_bf16(r2.x, r2.y); a.u[5] = pack2_bf16(r2.z, r2.w);
    a.u[6] = pack2_bf16(r3.x, r3.y); a.u[7] = pack2_bf16(r3.z, r3.w);

#pragma unroll
    for (int ct = 0; ct < 4; ++ct) {
      // B tile of W: two contiguous 16B loads from the swizzled table (L0/L2 hit)
      BV wb;
      const __bf16* wt = wl + (size_t)((kb >> 5) * 4 + ct) * 512;
      *reinterpret_cast<u4*>(&wb.u[0]) = *reinterpret_cast<const u4*>(wt);
      *reinterpret_cast<u4*>(&wb.u[4]) = *reinterpret_cast<const u4*>(wt + 8);
      acc[ct] = __builtin_amdgcn_wmma_f32_16x16x32_bf16(
          false, a.v, false, wb.v, (short)0, acc[ct], false, false);
    }
  }

  // bias + transposed bf16 store: lane holds col = ct*16+lr, rows m0+8g+0..7 contiguous
#pragma unroll
  for (int ct = 0; ct < 4; ++ct) {
    const int col = ct * 16 + lr;
    const float bv = bias[col];
    u4 st;
    st.x = pack2_bf16(acc[ct][0] + bv, acc[ct][1] + bv);
    st.y = pack2_bf16(acc[ct][2] + bv, acc[ct][3] + bv);
    st.z = pack2_bf16(acc[ct][4] + bv, acc[ct][5] + bv);
    st.w = pack2_bf16(acc[ct][6] + bv, acc[ct][7] + bv);
    __bf16* dptr = dst + (size_t)col * BTOT + m0 + 8 * g;
    *reinterpret_cast<u4*>(dptr) = st;  // 16B aligned
  }
}

// ---------------- Kernel 2: A = fx^T @ (fy.col(j) * fz), K-split partials ----------------
// Partial output layout (C-native): part[(j*8+ks)*4096 + ((m*4+t)*32 + L)*8 + r]
__global__ __launch_bounds__(256) void contract_kernel(
    const __bf16* __restrict__ fT, float* __restrict__ part) {
  const __bf16* __restrict__ fxT = fT;
  const __bf16* __restrict__ fyT = fT + (size_t)DDIM * BTOT;
  const __bf16* __restrict__ fzT = fT + (size_t)2 * DDIM * BTOT;

  const int j    = blockIdx.x;  // 0..63
  const int ks   = blockIdx.y;  // 0..7  (K split)
  const int tid  = threadIdx.x;
  const int wave = tid >> 5;
  const int lane = tid & 31;
  const int lr   = lane & 15;
  const int g    = lane >> 4;

  v8f acc[4][4] = {};  // [m-tile][k-tile], each 16x16 f32

  const int kb0 = ks * 8192 + wave * 1024;  // 32 steps of 32 per wave
  const __bf16* __restrict__ yp = fyT + (size_t)j * BTOT;

  for (int step = 0; step < 32; ++step) {
    const int kb = kb0 + step * 32;

    // A tiles (fx^T rows i, K=b): contiguous 16B runs
    BV a[4];
#pragma unroll
    for (int m = 0; m < 4; ++m) {
      const __bf16* ap = fxT + (size_t)(m * 16 + lr) * BTOT + kb + 16 * g;
      *reinterpret_cast<u4*>(&a[m].u[0]) = *reinterpret_cast<const u4*>(ap);
      *reinterpret_cast<u4*>(&a[m].u[4]) = *reinterpret_cast<const u4*>(ap + 8);
    }
    // fy column j: element e <-> b = kb + 16g + e (same for all lanes in a half)
    BV yv;
    const __bf16* yq = yp + kb + 16 * g;
    *reinterpret_cast<u4*>(&yv.u[0]) = *reinterpret_cast<const u4*>(yq);
    *reinterpret_cast<u4*>(&yv.u[4]) = *reinterpret_cast<const u4*>(yq + 8);

#pragma unroll
    for (int t = 0; t < 4; ++t) {
      // fz column k = t*16 + lr, then G = fy*fz with packed bf16 muls
      BV zv, gm;
      const __bf16* zq = fzT + (size_t)(t * 16 + lr) * BTOT + kb + 16 * g;
      *reinterpret_cast<u4*>(&zv.u[0]) = *reinterpret_cast<const u4*>(zq);
      *reinterpret_cast<u4*>(&zv.u[4]) = *reinterpret_cast<const u4*>(zq + 8);
#pragma unroll
      for (int q = 0; q < 8; ++q) gm.u[q] = pk_mul_bf16(yv.u[q], zv.u[q]);
#pragma unroll
      for (int m = 0; m < 4; ++m)
        acc[m][t] = __builtin_amdgcn_wmma_f32_16x16x32_bf16(
            false, a[m].v, false, gm.v, (short)0, acc[m][t], false, false);
    }
  }

  // Cross-wave reduction (8 waves -> 1) through 64 KB LDS with b128 DS ops.
  // Layout: lds[w][m][t][lane][r] -- each lane's 8-float accumulator is contiguous.
  __shared__ float lds[4][4][4][32][8];
  if (wave >= 4) {
#pragma unroll
    for (int m = 0; m < 4; ++m)
#pragma unroll
      for (int t = 0; t < 4; ++t)
        *reinterpret_cast<v8f*>(&lds[wave - 4][m][t][lane][0]) = acc[m][t];
  }
  __syncthreads();
  if (wave < 4) {
#pragma unroll
    for (int m = 0; m < 4; ++m)
#pragma unroll
      for (int t = 0; t < 4; ++t) {
        v8f* p = reinterpret_cast<v8f*>(&lds[wave][m][t][lane][0]);
        *p = *p + acc[m][t];  // packed f32 adds
      }
  }
  __syncthreads();

  // Dump reduced accumulators in C-native layout with contiguous 16B stores.
  float* __restrict__ outp = part + (size_t)(j * 8 + ks) * 4096;
  for (int c = tid; c < 512; c += 256) {  // c = (m*4+t)*32 + L, 2 per thread
    const int L = c & 31;
    const int t = (c >> 5) & 3;
    const int m = (c >> 7) & 3;
    v8f s = *reinterpret_cast<const v8f*>(&lds[0][m][t][L][0]) +
            *reinterpret_cast<const v8f*>(&lds[1][m][t][L][0]) +
            *reinterpret_cast<const v8f*>(&lds[2][m][t][L][0]) +
            *reinterpret_cast<const v8f*>(&lds[3][m][t][L][0]);
    *reinterpret_cast<v8f*>(outp + (size_t)c * 8) = s;
  }
}

// ---------------- Kernel 3: reduce K-splits, de-swizzle, scale by 1/B ----------------
__global__ __launch_bounds__(256) void reduce_kernel(
    const float* __restrict__ part, float* __restrict__ out) {
  const int o = blockIdx.x * 256 + threadIdx.x;  // i*4096 + j*64 + k
  const int i = o >> 12;
  const int j = (o >> 6) & 63;
  const int k = o & 63;
  // Invert the C-tile layout: i = m*16 + r + 8g, k = t*16 + N, L = g*16 + N
  const int m = i >> 4;
  const int r = i & 7;
  const int g = (i >> 3) & 1;
  const int t = k >> 4;
  const int N = k & 15;
  const int L = g * 16 + N;
  const size_t base = (size_t)((m * 4 + t) * 32 + L) * 8 + r;
  float s = 0.f;
#pragma unroll
  for (int ks = 0; ks < 8; ++ks)
    s += part[(size_t)(j * 8 + ks) * 4096 + base];
  out[o] = s * (1.0f / 65536.0f);
}

extern "C" void kernel_launch(void* const* d_in, const int* in_sizes, int n_in,
                              void* d_out, int out_size, void* d_ws, size_t ws_size,
                              hipStream_t stream) {
  (void)in_sizes; (void)n_in; (void)out_size; (void)ws_size;
  const float* x = (const float*)d_in[0];
  const float* y = (const float*)d_in[1];
  const float* z = (const float*)d_in[2];
  const float* W = (const float*)d_in[3];
  const float* b = (const float*)d_in[4];

  __bf16* fT  = (__bf16*)d_ws;                                        // 24 MB: fxT,fyT,fzT [64][65536] bf16
  float* part = (float*)((char*)d_ws + (size_t)3 * DDIM * BTOT * 2);  // 8 MB: 512 x [64x64] f32 partials
  __bf16* Wsw = (__bf16*)((char*)d_ws + (size_t)3 * DDIM * BTOT * 2 +
                          (size_t)512 * 4096 * 4);                    // 16 KB swizzled W

  wswz_kernel<<<32, 256, 0, stream>>>(W, Wsw);
  proj_kernel<<<dim3(512, 3), 256, 0, stream>>>(x, y, z, Wsw, b, fT);
  contract_kernel<<<dim3(64, 8), 256, 0, stream>>>(fT, part);
  reduce_kernel<<<1024, 256, 0, stream>>>(part, (float*)d_out);
}
